// BitSwiGLU_78245714199348
// MI455X (gfx1250) — 8-bit matrix operand order repaired and checked by reading, compiled, not run on hardware
//
#include <hip/hip_runtime.h>
#include <math.h>

typedef int v8i __attribute__((ext_vector_type(8)));

#define D_MODEL 2048
#define HIDDEN  8192
#define NTOK    8192
#define WCNT    16777216
#define LDSTR   80
#define TILEB   (64 * LDSTR)

#define ASYNC_TILE_LOAD_B128(ldsaddr, gptr)                                  \
  asm volatile("global_load_async_to_lds_b128 %0, %1, off" ::"v"(ldsaddr),   \
               "v"(gptr)                                                     \
               : "memory")
#define WAIT_ASYNC_0() asm volatile("s_wait_asynccnt 0" ::: "memory")

__device__ inline unsigned lds_addr_of(const void* p) {
  return (unsigned)(unsigned long long)p;
}

__global__ void zero_u32_kernel(unsigned int* __restrict__ p, int n) {
  int i = blockIdx.x * blockDim.x + threadIdx.x;
  if (i < n) p[i] = 0u;
}

__global__ void abssum_partial(const float* __restrict__ w,
                               float* __restrict__ partial, int n) {
  __shared__ float red[256];
  float s = 0.f;
  for (int i = blockIdx.x * blockDim.x + threadIdx.x; i < n;
       i += gridDim.x * blockDim.x)
    s += fabsf(w[i]);
  red[threadIdx.x] = s;
  __syncthreads();
  for (int o = 128; o > 0; o >>= 1) {
    if (threadIdx.x < o) red[threadIdx.x] += red[threadIdx.x + o];
    __syncthreads();
  }
  if (threadIdx.x == 0) partial[blockIdx.x] = red[0];
}

__global__ void finalize_gamma(const float* __restrict__ partial,
                               float* __restrict__ gamma) {
  __shared__ float red[256];
  for (int t = 0; t < 3; ++t) {
    float s = 0.f;
    for (int i = threadIdx.x; i < 2048; i += 256) s += partial[t * 2048 + i];
    red[threadIdx.x] = s;
    __syncthreads();
    for (int o = 128; o > 0; o >>= 1) {
      if (threadIdx.x < o) red[threadIdx.x] += red[threadIdx.x + o];
      __syncthreads();
    }
    if (threadIdx.x == 0) gamma[t] = fmaxf(red[0] / (float)WCNT, 1e-5f);
    __syncthreads();
  }
}

__global__ void quantize_w(const float* __restrict__ w,
                           const float* __restrict__ gamma, int gi,
                           signed char* __restrict__ wq, int n) {
  const float inv = 1.f / gamma[gi];
  int i = (blockIdx.x * blockDim.x + threadIdx.x) * 4;
  if (i < n) {
    float4 v = *(const float4*)(w + i);
    char4 q;
    q.x = (signed char)fminf(fmaxf(rintf(v.x * inv), -1.f), 1.f);
    q.y = (signed char)fminf(fmaxf(rintf(v.y * inv), -1.f), 1.f);
    q.z = (signed char)fminf(fmaxf(rintf(v.z * inv), -1.f), 1.f);
    q.w = (signed char)fminf(fmaxf(rintf(v.w * inv), -1.f), 1.f);
    *(char4*)(wq + i) = q;
  }
}

__global__ void quantize_x(const float* __restrict__ x,
                           signed char* __restrict__ xq,
                           float* __restrict__ sx) {
  __shared__ float red[256];
  const int row = blockIdx.x;
  const float* xr = x + (size_t)row * D_MODEL;
  float m = 0.f;
  for (int c = threadIdx.x * 4; c < D_MODEL; c += 1024) {
    float4 v = *(const float4*)(xr + c);
    m = fmaxf(m, fmaxf(fmaxf(fabsf(v.x), fabsf(v.y)),
                       fmaxf(fabsf(v.z), fabsf(v.w))));
  }
  red[threadIdx.x] = m;
  __syncthreads();
  for (int o = 128; o > 0; o >>= 1) {
    if (threadIdx.x < o)
      red[threadIdx.x] = fmaxf(red[threadIdx.x], red[threadIdx.x + o]);
    __syncthreads();
  }
  const float g = fmaxf(red[0], 1e-5f);
  const float qs = 127.f / g;
  if (threadIdx.x == 0) sx[row] = g / 127.f;
  signed char* xqr = xq + (size_t)row * D_MODEL;
  for (int c = threadIdx.x * 4; c < D_MODEL; c += 1024) {
    float4 v = *(const float4*)(xr + c);
    char4 q;
    q.x = (signed char)fminf(fmaxf(rintf(v.x * qs), -128.f), 127.f);
    q.y = (signed char)fminf(fmaxf(rintf(v.y * qs), -128.f), 127.f);
    q.z = (signed char)fminf(fmaxf(rintf(v.z * qs), -128.f), 127.f);
    q.w = (signed char)fminf(fmaxf(rintf(v.w * qs), -128.f), 127.f);
    *(char4*)(xqr + c) = q;
  }
}

__global__ void quantize_h(const float* __restrict__ h,
                           const unsigned int* __restrict__ rowabs,
                           signed char* __restrict__ hq,
                           float* __restrict__ sh) {
  const int row = blockIdx.x;
  const float g = fmaxf(__uint_as_float(rowabs[row]), 1e-5f);
  const float qs = 127.f / g;
  if (threadIdx.x == 0) sh[row] = g / 127.f;
  const float* hr = h + (size_t)row * HIDDEN;
  signed char* hqr = hq + (size_t)row * HIDDEN;
  for (int c = threadIdx.x * 4; c < HIDDEN; c += 1024) {
    float4 v = *(const float4*)(hr + c);
    char4 q;
    q.x = (signed char)fminf(fmaxf(rintf(v.x * qs), -128.f), 127.f);
    q.y = (signed char)fminf(fmaxf(rintf(v.y * qs), -128.f), 127.f);
    q.z = (signed char)fminf(fmaxf(rintf(v.z * qs), -128.f), 127.f);
    q.w = (signed char)fminf(fmaxf(rintf(v.w * qs), -128.f), 127.f);
    *(char4*)(hqr + c) = q;
  }
}

__device__ inline v8i load_afrag(const signed char* base) {
  int2 p0 = *(const int2*)(base + 0);
  int2 p1 = *(const int2*)(base + 16);
  int2 p2 = *(const int2*)(base + 32);
  int2 p3 = *(const int2*)(base + 48);
  v8i a;
  a[0] = p0.x; a[1] = p0.y; a[2] = p1.x; a[3] = p1.y;
  a[4] = p2.x; a[5] = p2.y; a[6] = p3.x; a[7] = p3.y;
  return a;
}
__device__ inline v8i load_bfrag(const signed char* base) {
  int2 p0 = *(const int2*)(base + 0);
  int2 p1 = *(const int2*)(base + 16);
  int2 p2 = *(const int2*)(base + 32);
  int2 p3 = *(const int2*)(base + 48);
  v8i b;
  b[0] = p0.x; b[1] = p0.y; b[2] = p1.x; b[3] = p1.y;
  b[4] = p2.x; b[5] = p2.y; b[6] = p3.x; b[7] = p3.y;
  return b;
}

__device__ inline v8i wmma_iu8(v8i a, v8i b, v8i c) {
  return __builtin_amdgcn_wmma_i32_16x16x64_iu8(true, a, true, b, c, false, false);
}

__global__ __launch_bounds__(256) void gemm_gateval(
    const signed char* __restrict__ xq, const float* __restrict__ sx,
    const signed char* __restrict__ wqg, const signed char* __restrict__ wqv,
    const float* __restrict__ gate_b, const float* __restrict__ val_b,
    const float* __restrict__ gamma, float* __restrict__ h,
    unsigned int* __restrict__ rowabs) {
  __shared__ signed char As[2][TILEB];
  __shared__ signed char Bg[2][TILEB];
  __shared__ signed char Bv[2][TILEB];
  __shared__ unsigned int rmax[64];

  const int tid  = threadIdx.x;
  const int lane = tid & 31;
  const int wave = tid >> 5;
  const int m0 = (wave & 3) << 4;
  const int n0 = (wave >> 2) << 5;
  const long blockM = (long)blockIdx.x * 64;
  const long blockN = (long)blockIdx.y * 64;

  if (tid < 64) rmax[tid] = 0u;

  const int ldRow = tid >> 2;
  const int ldOff = (tid & 3) << 4;
  const unsigned stOff = (unsigned)(ldRow * LDSTR + ldOff);
  const signed char* gA = xq  + (blockM + ldRow) * (long)D_MODEL + ldOff;
  const signed char* gG = wqg + (blockN + ldRow) * (long)D_MODEL + ldOff;
  const signed char* gV = wqv + (blockN + ldRow) * (long)D_MODEL + ldOff;

  const unsigned lA0 = lds_addr_of(&As[0][0]) + stOff;
  const unsigned lG0 = lds_addr_of(&Bg[0][0]) + stOff;
  const unsigned lV0 = lds_addr_of(&Bv[0][0]) + stOff;

  ASYNC_TILE_LOAD_B128(lA0, gA);
  ASYNC_TILE_LOAD_B128(lG0, gG);
  ASYNC_TILE_LOAD_B128(lV0, gV);

  v8i accg0 = {}, accg1 = {}, accv0 = {}, accv1 = {};

  const int arowOff = (m0 + (lane & 15)) * LDSTR + ((lane >> 4) << 3);
  const int b0Off = (n0 + (lane & 15)) * LDSTR + ((lane >> 4) << 3);
  const int b1Off = b0Off + (16 * LDSTR);

  const int NSTEP = D_MODEL / 64;
#pragma clang loop unroll(disable)
  for (int s = 0; s < NSTEP; ++s) {
    const unsigned sel = (s & 1) ? (unsigned)TILEB : 0u;
    WAIT_ASYNC_0();
    __syncthreads();
    if (s + 1 < NSTEP) {
      const unsigned osel = (unsigned)TILEB - sel;
      const int k = (s + 1) * 64;
      ASYNC_TILE_LOAD_B128(lA0 + osel, gA + k);
      ASYNC_TILE_LOAD_B128(lG0 + osel, gG + k);
      ASYNC_TILE_LOAD_B128(lV0 + osel, gV + k);
    }
    const signed char* Ab = &As[0][0] + sel;
    const signed char* Gb = &Bg[0][0] + sel;
    const signed char* Vb = &Bv[0][0] + sel;
    const v8i a = load_afrag(Ab + arowOff);
    v8i bg0 = load_bfrag(Gb + b0Off);
    v8i bv0 = load_bfrag(Vb + b0Off);
    v8i bg1 = load_bfrag(Gb + b1Off);
    v8i bv1 = load_bfrag(Vb + b1Off);
    accg0 = wmma_iu8(a, bg0, accg0);
    accv0 = wmma_iu8(a, bv0, accv0);
    accg1 = wmma_iu8(a, bg1, accg1);
    accv1 = wmma_iu8(a, bv1, accv1);
  }

  const float gammaG = gamma[0];
  const float gammaV = gamma[1];
#pragma unroll
  for (int t = 0; t < 2; ++t) {
    const v8i ag = t ? accg1 : accg0;
    const v8i av = t ? accv1 : accv0;
    const long col = blockN + n0 + (t << 4) + (lane & 15);
    const float bG = gate_b[col];
    const float bV = val_b[col];
#pragma unroll
    for (int r = 0; r < 8; ++r) {
      const int mloc = m0 + r + ((lane >> 4) << 3);
      const long row = blockM + mloc;
      const float sxr = sx[row];
      const float gate = (float)ag[r] * sxr * gammaG + bG;
      const float val  = (float)av[r] * sxr * gammaV + bV;
      const float hv = (gate / (1.f + __expf(-gate))) * val;
      h[row * (long)HIDDEN + col] = hv;
      atomicMax(&rmax[mloc], __float_as_uint(fabsf(hv)));
    }
  }
  __syncthreads();
  if (tid < 64) atomicMax(&rowabs[blockM + tid], rmax[tid]);
}

__global__ __launch_bounds__(256) void gemm_out_k(
    const signed char* __restrict__ hq, const float* __restrict__ sh,
    const signed char* __restrict__ wqo, const float* __restrict__ out_b,
    const float* __restrict__ gamma, float* __restrict__ out) {
  __shared__ signed char As[2][TILEB];
  __shared__ signed char Bs[2][TILEB];

  const int tid  = threadIdx.x;
  const int lane = tid & 31;
  const int wave = tid >> 5;
  const int m0 = (wave & 3) << 4;
  const int n0 = (wave >> 2) << 5;
  const long blockM = (long)blockIdx.x * 64;
  const long blockN = (long)blockIdx.y * 64;

  const int ldRow = tid >> 2;
  const int ldOff = (tid & 3) << 4;
  const unsigned stOff = (unsigned)(ldRow * LDSTR + ldOff);
  const signed char* gA = hq  + (blockM + ldRow) * (long)HIDDEN + ldOff;
  const signed char* gB = wqo + (blockN + ldRow) * (long)HIDDEN + ldOff;

  const unsigned lA0 = lds_addr_of(&As[0][0]) + stOff;
  const unsigned lB0 = lds_addr_of(&Bs[0][0]) + stOff;

  ASYNC_TILE_LOAD_B128(lA0, gA);
  ASYNC_TILE_LOAD_B128(lB0, gB);

  v8i acc0 = {}, acc1 = {};

  const int arowOff = (m0 + (lane & 15)) * LDSTR + ((lane >> 4) << 3);
  const int b0Off = (n0 + (lane & 15)) * LDSTR + ((lane >> 4) << 3);
  const int b1Off = b0Off + (16 * LDSTR);

  const int NSTEP = HIDDEN / 64;
#pragma clang loop unroll(disable)
  for (int s = 0; s < NSTEP; ++s) {
    const unsigned sel = (s & 1) ? (unsigned)TILEB : 0u;
    WAIT_ASYNC_0();
    __syncthreads();
    if (s + 1 < NSTEP) {
      const unsigned osel = (unsigned)TILEB - sel;
      const int k = (s + 1) * 64;
      ASYNC_TILE_LOAD_B128(lA0 + osel, gA + k);
      ASYNC_TILE_LOAD_B128(lB0 + osel, gB + k);
    }
    const signed char* Ab = &As[0][0] + sel;
    const signed char* Bb = &Bs[0][0] + sel;
    const v8i a = load_afrag(Ab + arowOff);
    v8i b0 = load_bfrag(Bb + b0Off);
    v8i b1 = load_bfrag(Bb + b1Off);
    acc0 = wmma_iu8(a, b0, acc0);
    acc1 = wmma_iu8(a, b1, acc1);
  }

  const float gammaO = gamma[2];
#pragma unroll
  for (int t = 0; t < 2; ++t) {
    const v8i ac = t ? acc1 : acc0;
    const long col = blockN + n0 + (t << 4) + (lane & 15);
    const float bO = out_b[col];
#pragma unroll
    for (int r = 0; r < 8; ++r) {
      const long row = blockM + m0 + r + ((lane >> 4) << 3);
      out[row * (long)D_MODEL + col] = (float)ac[r] * sh[row] * gammaO + bO;
    }
  }
}

extern "C" void kernel_launch(void* const* d_in, const int* in_sizes, int n_in,
                              void* d_out, int out_size, void* d_ws,
                              size_t ws_size, hipStream_t stream) {
  (void)in_sizes; (void)n_in; (void)out_size; (void)ws_size;
  const float* x      = (const float*)d_in[0];
  const float* gate_w = (const float*)d_in[1];
  const float* gate_b = (const float*)d_in[2];
  const float* val_w  = (const float*)d_in[3];
  const float* val_b  = (const float*)d_in[4];
  const float* out_w  = (const float*)d_in[5];
  const float* out_b  = (const float*)d_in[6];
  float* out = (float*)d_out;

  char* ws = (char*)d_ws;
  size_t off = 0;
  auto take = [&](size_t bytes) -> char* {
    char* p = ws + off;
    off = (off + bytes + 255) & ~((size_t)255);
    return p;
  };

  float*        partial = (float*)take(3 * 2048 * sizeof(float));
  float*        gamma   = (float*)take(16 * sizeof(float));
  signed char*  xq      = (signed char*)take((size_t)NTOK * D_MODEL);
  float*        sx      = (float*)take((size_t)NTOK * sizeof(float));
  signed char*  wqg     = (signed char*)take((size_t)HIDDEN * D_MODEL);
  signed char*  wqv     = (signed char*)take((size_t)HIDDEN * D_MODEL);
  signed char*  wqo     = (signed char*)take((size_t)D_MODEL * HIDDEN);
  unsigned int* rowabs  = (unsigned int*)take((size_t)NTOK * sizeof(unsigned int));
  float*        sh      = (float*)take((size_t)NTOK * sizeof(float));
  signed char*  hq      = (signed char*)take((size_t)NTOK * HIDDEN);
  float*        hbuf    = (float*)take((size_t)NTOK * HIDDEN * sizeof(float));

  zero_u32_kernel<<<(NTOK + 255) / 256, 256, 0, stream>>>(rowabs, NTOK);

  abssum_partial<<<2048, 256, 0, stream>>>(gate_w, partial + 0 * 2048, WCNT);
  abssum_partial<<<2048, 256, 0, stream>>>(val_w,  partial + 1 * 2048, WCNT);
  abssum_partial<<<2048, 256, 0, stream>>>(out_w,  partial + 2 * 2048, WCNT);
  finalize_gamma<<<1, 256, 0, stream>>>(partial, gamma);

  quantize_w<<<WCNT / 1024, 256, 0, stream>>>(gate_w, gamma, 0, wqg, WCNT);
  quantize_w<<<WCNT / 1024, 256, 0, stream>>>(val_w,  gamma, 1, wqv, WCNT);
  quantize_w<<<WCNT / 1024, 256, 0, stream>>>(out_w,  gamma, 2, wqo, WCNT);

  quantize_x<<<NTOK, 256, 0, stream>>>(x, xq, sx);

  gemm_gateval<<<dim3(NTOK / 64, HIDDEN / 64), 256, 0, stream>>>(
      xq, sx, wqg, wqv, gate_b, val_b, gamma, hbuf, rowabs);

  quantize_h<<<NTOK, 256, 0, stream>>>(hbuf, rowabs, hq, sh);

  gemm_out_k<<<dim3(NTOK / 64, D_MODEL / 64), 256, 0, stream>>>(
      hq, sh, wqo, out_b, gamma, out);
}
